// MeanPoolGraphSage_79972291052243
// MI455X (gfx1250) — matrix operand order repaired and checked by reading, not compiled, not run on hardware
//
#include <hip/hip_runtime.h>
#include <hip/hip_bf16.h>

// ---------------------------------------------------------------------------
// MeanPoolGraphSage on MI455X (gfx1250, wave32, WMMA + TDM)
//
// d_ws layout: summed[N*256] f32 | counts[N] f32 | wt[256][128] f16 (W1^T)
//              | skT[64][128] f16 (self_kernel^T) | nkT[64][256] f16 (neigh^T)
// ---------------------------------------------------------------------------

typedef __attribute__((ext_vector_type(16))) _Float16 v16h;
typedef __attribute__((ext_vector_type(8)))  float    v8f;

union Frag {            // 8 dwords == 16 halves == one WMMA A/B operand
    unsigned int u[8];
    v16h h;
};

#define F_IN    128
#define HID     256
#define KU      64
#define UNITS   128

#if defined(__gfx1250__) && __has_builtin(__builtin_amdgcn_tensor_load_to_lds)
#define HAVE_TDM 1
#else
#define HAVE_TDM 0
#endif

#if HAVE_TDM
typedef __attribute__((ext_vector_type(4))) unsigned int v4u_;
typedef __attribute__((ext_vector_type(8))) int          v8i_;
typedef __attribute__((ext_vector_type(4))) int          v4i_;

// 2D f16 tile load via Tensor Data Mover. LDS destination gets padded rows:
// pad_interval_code selects 2^(code+1) dwords between pads, pad_amount=1 dword.
__device__ __forceinline__ void tdm_load_tile_f16(
    unsigned lds_off, const _Float16* gsrc,
    unsigned tensor_d0, unsigned tensor_d1,
    unsigned tile_d0, unsigned tile_d1,
    unsigned pad_interval_code)
{
    unsigned long long ga = (unsigned long long)(const void*)gsrc;
    v4u_ g0;
    g0[0] = 1u;                                         // count = 1 (valid D#)
    g0[1] = lds_off;                                    // lds_addr
    g0[2] = (unsigned)ga;                               // global_addr[31:0]
    g0[3] = (unsigned)((ga >> 32) & 0x1FFFFFFull)       // global_addr[56:32]
          | (2u << 30);                                 // type = 2 ("image")

    unsigned long long d1s = (unsigned long long)tensor_d0 * tensor_d1;
    v8i_ g1;
    g1[0] = (int)((1u << 16)                            // data_size = 2 bytes
                | (1u << 20)                            // pad_enable
                | (pad_interval_code << 22));           // pad_amount = 0 (1 dw)
    g1[1] = (int)((tensor_d0 & 0xFFFFu) << 16);         // tensor_dim0[15:0]
    g1[2] = (int)((tensor_d0 >> 16) | ((tensor_d1 & 0xFFFFu) << 16));
    g1[3] = (int)((tensor_d1 >> 16) | (tile_d0 << 16)); // tile_dim0
    g1[4] = (int)(tile_d1 & 0xFFFFu);                   // tile_dim1, tile_dim2=0
    g1[5] = (int)tensor_d0;                             // dim0 stride [31:0]
    g1[6] = (int)(((unsigned)(d1s & 0xFFFFull)) << 16); // dim1 stride [15:0]
    g1[7] = (int)(d1s >> 16);                           // dim1 stride [47:16]

    v4i_ z4 = {0, 0, 0, 0};
#if __clang_major__ >= 23
    v8i_ z8 = {0, 0, 0, 0, 0, 0, 0, 0};
    __builtin_amdgcn_tensor_load_to_lds(g0, g1, z4, z4, z8, 0);
#else
    __builtin_amdgcn_tensor_load_to_lds(g0, g1, z4, z4, 0);
#endif
}
#endif // HAVE_TDM

// -------------------------------- zero workspace ---------------------------
__global__ void zero_ws_kernel(float* __restrict__ p, long long n) {
    long long i = (long long)blockIdx.x * blockDim.x + threadIdx.x;
    long long stride = (long long)gridDim.x * blockDim.x;
    for (; i < n; i += stride) p[i] = 0.0f;
}

// --------------------- convert weights to f16, transposed ------------------
// wt[n][k]  = w1[k][n]        (256 x 128)
// skT[n][k] = self_k[k][n]    (64 x 128)
// nkT[n][k] = neigh_k[k][n]   (64 x 256)
__global__ void convert_weights_kernel(const float* __restrict__ w1,
                                       const float* __restrict__ sk,
                                       const float* __restrict__ nk,
                                       _Float16* __restrict__ wt,
                                       _Float16* __restrict__ skT,
                                       _Float16* __restrict__ nkT) {
    int i = blockIdx.x * 256 + threadIdx.x;
    if (i < 256 * 128) {
        int n = i >> 7, k = i & 127;
        wt[i] = (_Float16)w1[(size_t)k * HID + n];
    } else if (i < 256 * 128 + 64 * 128) {
        int j = i - 256 * 128;
        int n = j >> 7, k = j & 127;
        skT[j] = (_Float16)sk[(size_t)k * KU + n];
    } else if (i < 256 * 128 + 64 * 128 + 64 * 256) {
        int j = i - (256 * 128 + 64 * 128);
        int n = j >> 8, k = j & 255;
        nkT[j] = (_Float16)nk[(size_t)k * KU + n];
    }
}

// -------------------------------- edge MLP + scatter -----------------------
// 64 edges per block, 256 threads (8 waves).
// Per wave: 8 output tiles (2 N-tiles x 4 A-tiles), 4 K-chunks of 32.
// W1 chunks stream in via TDM, double buffered, overlapped with WMMA compute.
#define EPB   64
#define ASTR  136   // LDS stride (halves) for A tiles, 272B rows (4B aligned)
#define WSTR  34    // LDS stride (halves) for W1 chunk (TDM pad: 16dw + 1dw)
#if HAVE_TDM
#define WBUFS 2
#else
#define WBUFS 1
#endif

__global__ void __launch_bounds__(256)
edge_mlp_kernel(const float* __restrict__ x,
                const int*   __restrict__ row,
                const int*   __restrict__ col,
                const float* __restrict__ ew,
                const _Float16* __restrict__ wt, // W1^T [256][128] f16
                const float* __restrict__ b1,    // [256]
                float* __restrict__ summed,      // [N][256]
                float* __restrict__ counts,      // [N]
                int nEdges)
{
    __shared__ _Float16 Ash[EPB][ASTR];          // gathered features, f16
    __shared__ _Float16 Wsh[WBUFS][HID][WSTR];   // W1 chunk(s): [n][kk]
    __shared__ int      rowsh[EPB];

    const int tid  = threadIdx.x;
    const int e0   = blockIdx.x * EPB;
    const int lane = tid & 31;
    const int wv   = tid >> 5;

#if HAVE_TDM
    // kick off chunk 0 immediately; it streams while we gather features
    if (wv == 0) {
        tdm_load_tile_f16((unsigned)(unsigned long long)(void*)&Wsh[0][0][0],
                          wt /* tile start: col 0 */, F_IN, HID,
                          /*tile*/ 32, HID, /*pad code 16dw*/ 3);
    }
#endif

    // stage A: gather x[col[e]] * ew[e], convert to f16
    for (int idx = tid; idx < EPB * F_IN; idx += 256) {
        int e = idx >> 7, f = idx & 127;
        int ge = e0 + e;
        float v = 0.0f;
        if (ge < nEdges) v = x[(size_t)col[ge] * F_IN + f] * ew[ge];
        Ash[e][f] = (_Float16)v;
    }
    if (tid < EPB) {
        int ge = e0 + tid;
        if (ge < nEdges) {
            int r = row[ge];
            rowsh[tid] = r;
            atomicAdd(&counts[r], 1.0f);
        } else {
            rowsh[tid] = -1;
        }
    }
    __syncthreads();

    v8f acc[8] = {};                             // 8 persistent 16x16 f32 tiles
    const int mA  = lane & 15;
    const int kbA = (lane >> 4) * 8;
    const int nB  = lane & 15;
    const int koB = (lane >> 4) * 8;

    for (int kc = 0; kc < 4; ++kc) {
#if HAVE_TDM
        if (wv == 0) __builtin_amdgcn_s_wait_tensorcnt(0);  // chunk kc landed
        __syncthreads();
        if (wv == 0 && kc < 3) {                 // prefetch next chunk
            tdm_load_tile_f16(
                (unsigned)(unsigned long long)(void*)&Wsh[(kc + 1) & 1][0][0],
                wt + (kc + 1) * 32, F_IN, HID, 32, HID, 3);
        }
        const _Float16 (*W)[WSTR] = Wsh[kc & 1];
#else
        __syncthreads();
        for (int idx = tid; idx < HID * 32; idx += 256) {
            int n = idx >> 5, kk = idx & 31;
            Wsh[0][n][kk] = wt[(size_t)n * F_IN + kc * 32 + kk];
        }
        __syncthreads();
        const _Float16 (*W)[WSTR] = Wsh[0];
#endif

        Frag bf;
#pragma unroll
        for (int j = 0; j < 8; ++j) {
            const int jobid = wv * 8 + j;
            const int atile = jobid & 3;
            const int ntile = jobid >> 2;
            if ((j & 3) == 0) {                  // B frag reused across 4 A-tiles
                const int n = ntile * 16 + nB;
#pragma unroll
                for (int v = 0; v < 8; ++v)
                    bf.u[v] = *(const unsigned int*)&W[n][koB + (v < 4 ? 2 * v : 16 + 2 * (v - 4))];
            }
            Frag af;
            const int m = atile * 16 + mA;
#pragma unroll
            for (int v = 0; v < 8; ++v) {
                const int k = kc * 32 + kbA + (v < 4 ? 2 * v : 16 + 2 * (v - 4));
                af.u[v] = *(const unsigned int*)&Ash[m][k];
            }
            acc[j] = __builtin_amdgcn_wmma_f32_16x16x32_f16(
                false, af.h, false, bf.h, (short)0, acc[j], false, false);
        }
    }

    // epilogue: bias + relu + atomic scatter into summed[row]
#pragma unroll
    for (int j = 0; j < 8; ++j) {
        const int jobid = wv * 8 + j;
        const int atile = jobid & 3;
        const int ntile = jobid >> 2;
        const int n = ntile * 16 + (lane & 15);
        const float bn = b1[n];
        const int mbase = atile * 16 + ((lane >> 4) ? 8 : 0);
#pragma unroll
        for (int r = 0; r < 8; ++r) {
            const int rid = rowsh[mbase + r];
            if (rid >= 0) {
                float h = acc[j][r] + bn;
                h = h > 0.0f ? h : 0.0f;
                atomicAdd(&summed[(size_t)rid * HID + n], h);
            }
        }
    }
}

// -------------------------------- node-side GEMMs + concat -----------------
// 16 nodes per block, 256 threads (8 waves).
// waves 0..3: from_x = x @ self_kernel       (K=128, N-tile = wave)
// waves 4..7: from_n = mean_h @ neigh_kernel (K=256, N-tile = wave-4)
// Both weight matrices land in LDS via TDM (overlapped with A staging).
#define XSTR 136
#define HSTR 264
#define SKSTR 130   // 64dw rows + 1dw TDM pad
#define NKSTR 258   // 128dw rows + 1dw TDM pad

__global__ void __launch_bounds__(256)
node_out_kernel(const float* __restrict__ x,
                const float* __restrict__ summed,
                const float* __restrict__ counts,
                const _Float16* __restrict__ skT,  // [64][128] f16
                const _Float16* __restrict__ nkT,  // [64][256] f16
                const float* __restrict__ bias,    // [128]
                float* __restrict__ out,           // [N][128]
                int nNodes)
{
    __shared__ _Float16 xa[16][XSTR];       //  4352 B
    __shared__ _Float16 ha[16][HSTR];       //  8448 B
    __shared__ _Float16 skLds[KU][SKSTR];   // 16640 B
    __shared__ _Float16 nkLds[KU][NKSTR];   // 33024 B
    __shared__ float    invc[16];

    const int tid  = threadIdx.x;
    const int n0   = blockIdx.x * 16;
    const int lane = tid & 31;
    const int wv   = tid >> 5;

#if HAVE_TDM
    if (wv == 0) {   // stream both weight matrices while we stage A operands
        tdm_load_tile_f16((unsigned)(unsigned long long)(void*)&skLds[0][0],
                          skT, F_IN, KU, F_IN, KU, /*64dw*/ 5);
        tdm_load_tile_f16((unsigned)(unsigned long long)(void*)&nkLds[0][0],
                          nkT, HID, KU, HID, KU, /*128dw*/ 6);
    }
#else
    for (int idx = tid; idx < KU * F_IN; idx += 256) {
        int n = idx >> 7, k = idx & 127;
        skLds[n][k] = skT[idx];
    }
    for (int idx = tid; idx < KU * HID; idx += 256) {
        int n = idx >> 8, k = idx & 255;
        nkLds[n][k] = nkT[idx];
    }
#endif

    if (tid < 16) {
        float c = counts[n0 + tid];
        invc[tid] = 1.0f / fmaxf(c, 1.0f);
    }
    __syncthreads();

    for (int idx = tid; idx < 16 * F_IN; idx += 256) {
        int i = idx >> 7, f = idx & 127;
        xa[i][f] = (_Float16)x[(size_t)(n0 + i) * F_IN + f];
    }
    for (int idx = tid; idx < 16 * HID; idx += 256) {
        int i = idx >> 8, c = idx & 255;
        ha[i][c] = (_Float16)(summed[(size_t)(n0 + i) * HID + c] * invc[i]);
    }
#if HAVE_TDM
    if (wv == 0) __builtin_amdgcn_s_wait_tensorcnt(0);
#endif
    __syncthreads();

    const bool g2 = (wv >= 4);
    const _Float16* Bls  = g2 ? &nkLds[0][0] : &skLds[0][0];
    const int       bstr = g2 ? NKSTR : SKSTR;
    const _Float16* Als  = g2 ? &ha[0][0] : &xa[0][0];
    const int       astr = g2 ? HSTR : XSTR;
    const int       KC   = g2 ? 8 : 4;           // K-chunks of 32
    const int       nt   = wv & 3;

    const int mA   = lane & 15;
    const int kbA  = (lane >> 4) * 8;
    const int ncol = nt * 16 + (lane & 15);

    v8f acc = {};
    for (int kc = 0; kc < KC; ++kc) {
        Frag bf;
        const int koff = kc * 32 + (lane >> 4) * 8;
#pragma unroll
        for (int v = 0; v < 8; ++v)
            bf.u[v] = *(const unsigned int*)&Bls[(size_t)ncol * bstr + koff + (v < 4 ? 2 * v : 16 + 2 * (v - 4))];
        Frag af;
#pragma unroll
        for (int v = 0; v < 8; ++v) {
            const int k = kc * 32 + kbA + (v < 4 ? 2 * v : 16 + 2 * (v - 4));
            af.u[v] = *(const unsigned int*)&Als[(size_t)mA * astr + k];
        }
        acc = __builtin_amdgcn_wmma_f32_16x16x32_f16(
            false, af.h, false, bf.h, (short)0, acc, false, false);
    }

    // concat([from_x, from_neighs]) + bias + relu
    const int obase = (g2 ? KU : 0) + nt * 16 + (lane & 15);
    const float bb = bias[obase];
    const int moff = (lane >> 4) ? 8 : 0;
#pragma unroll
    for (int r = 0; r < 8; ++r) {
        const int node = n0 + moff + r;
        if (node < nNodes) {
            float o = acc[r] + bb;
            o = o > 0.0f ? o : 0.0f;
            out[(size_t)node * UNITS + obase] = o;
        }
    }
}

// ---------------------------------------------------------------------------
extern "C" void kernel_launch(void* const* d_in, const int* in_sizes, int n_in,
                              void* d_out, int out_size, void* d_ws, size_t ws_size,
                              hipStream_t stream) {
    const float* x       = (const float*)d_in[0];
    const int*   ei      = (const int*)  d_in[1];   // [2][E] flat
    const float* ew      = (const float*)d_in[2];
    const float* self_k  = (const float*)d_in[3];
    const float* w1      = (const float*)d_in[4];
    const float* b1      = (const float*)d_in[5];
    const float* neigh_k = (const float*)d_in[6];
    const float* bias    = (const float*)d_in[7];
    float*       out     = (float*)d_out;

    const int E = in_sizes[2];
    const int N = in_sizes[0] / F_IN;

    float*    summed = (float*)d_ws;                     // [N][256]
    float*    counts = summed + (size_t)N * HID;         // [N]
    _Float16* wt     = (_Float16*)(counts + N);          // [256][128]
    _Float16* skT    = wt + 256 * 128;                   // [64][128]
    _Float16* nkT    = skT + 64 * 128;                   // [64][256]

    const int* row = ei;
    const int* col = ei + E;

    const long long zn = (long long)N * HID + N;
    zero_ws_kernel<<<2048, 256, 0, stream>>>(summed, zn);

    const int cvt_elems = 256 * 128 + 64 * 128 + 64 * 256;
    convert_weights_kernel<<<(cvt_elems + 255) / 256, 256, 0, stream>>>(
        w1, self_k, neigh_k, wt, skT, nkT);

    edge_mlp_kernel<<<(E + EPB - 1) / EPB, 256, 0, stream>>>(
        x, row, col, ew, wt, b1, summed, counts, E);

    node_out_kernel<<<(N + 15) / 16, 256, 0, stream>>>(
        x, summed, counts, skT, nkT, bias, out, N);
}
